// AdaptiveMetaLearnerV1_91147795956325
// MI455X (gfx1250) — compile-verified, matrix operand order checked by reading, not run on hardware
//
#include <hip/hip_runtime.h>

#define NPAR 500000
#define HH 20
#define G4 80
#define NLAYERS 2
#define EPSV 1e-5f
#define NW 3                       // waves per block
#define TPB (NW * 32)
#define NTILES (NPAR / 32)         // 15625 (exact)
#define NBLK 2048
#define WROW 32                    // padded K per cached row (zeros at 20..31 -> maskless frags)

typedef __attribute__((ext_vector_type(16))) _Float16 v16h;
typedef __attribute__((ext_vector_type(8)))  _Float16 v8h;
typedef __attribute__((ext_vector_type(4)))  _Float16 v4h;
typedef __attribute__((ext_vector_type(8)))  float    v8f;

struct BranchP {
  const float *lin1_w, *lin1_b, *ln1_g, *ln1_b, *wi, *wh;
  const float *lgi_g, *lgi_b, *lgh_g, *lgh_b, *lnc_g, *lnc_b, *out_w, *out_b;
};
struct KParams {
  const float *x, *hx0, *cx0, *act_hx0, *act_cx0;
  BranchP bp[2];
  float* xout;     // d_out (N floats of x_out; d_out[N] = qt_out via reduce kernel)
  float* partial;  // d_ws per-block partials
};

__device__ __forceinline__ float rcp_(float x)  { return __builtin_amdgcn_rcpf(x); }
__device__ __forceinline__ float sqrt_(float x) { return __builtin_amdgcn_sqrtf(x); }
__device__ __forceinline__ float sigmoid_(float x) { return rcp_(1.f + __expf(-x)); }
__device__ __forceinline__ float tanh_(float x) {
  float e = __expf(-2.f * fabsf(x));           // v_exp_f32 (TRANS, co-executes with WMMA)
  float t = (1.f - e) * rcp_(1.f + e);
  return copysignf(t, x);
}

__device__ __forceinline__ float hsum16(float v) {
  v += __shfl_xor(v, 1, 32);
  v += __shfl_xor(v, 2, 32);
  v += __shfl_xor(v, 4, 32);
  v += __shfl_xor(v, 8, 32);
  return v;
}

// Wave-local LDS producer->consumer ordering (CDNA5 split counters).
__device__ __forceinline__ void wave_lds_sync() {
  __builtin_amdgcn_wave_barrier();
  asm volatile("s_wait_dscnt 0" ::: "memory");
  __builtin_amdgcn_wave_barrier();
}

__device__ __forceinline__ v8f wmma16(v16h a, v16h b, v8f c) {
  // D = A(16x32 f16) * B(32x16 f16) + C(16x16 f32)
  return __builtin_amdgcn_wmma_f32_16x16x32_f16(false, a, false, b, (short)0, c, false, false);
}

// Maskless fragment from an LDS row of 32 padded halves (zeros at K=20..31).
// lane<16 (k0=0): K=0..7 and K=16..23 ; lane>=16 (k0=8): K=8..15 and K=24..31(=0).
__device__ __forceinline__ v16h frag_from_lds(const _Float16* row, int k0) {
  union { v16h v; v8h h[2]; } u;
  u.h[0] = *(const v8h*)(row + k0);
  u.h[1] = *(const v8h*)(row + k0 + 16);
  return u.v;
}

// Fragment (row/col per lane) from 20 contiguous global f32, K zero-padded to 32.
__device__ __forceinline__ v16h frag_from_global(const float* wrow, int k0, bool lo) {
  float4 f0 = *(const float4*)(wrow + k0);
  float4 f1 = *(const float4*)(wrow + k0 + 4);
  float4 f2 = *(const float4*)(wrow + 16);     // always valid
  float zf = lo ? 1.f : 0.f;
  v16h a;
  a[0] = (_Float16)f0.x; a[1] = (_Float16)f0.y; a[2] = (_Float16)f0.z; a[3] = (_Float16)f0.w;
  a[4] = (_Float16)f1.x; a[5] = (_Float16)f1.y; a[6] = (_Float16)f1.z; a[7] = (_Float16)f1.w;
  a[8]  = (_Float16)(f2.x * zf);
  a[9]  = (_Float16)(f2.y * zf);
  a[10] = (_Float16)(f2.z * zf);
  a[11] = (_Float16)(f2.w * zf);
  _Float16 z = (_Float16)0.f;
  a[12] = z; a[13] = z; a[14] = z; a[15] = z;
  return a;
}

__global__ void __launch_bounds__(TPB)
aml_main(KParams P)
{
  // 8 weight matrices (b,L,{wi,wh}) x 80 cols x 32 padded K, f16: 40 KB
  __shared__ __align__(16) _Float16 wbuf[8 * G4][WROW];
  __shared__ __align__(16) _Float16 hbuf[NW][32][WROW];   // 6 KB
  __shared__ __align__(16) _Float16 gbuf[NW][32][G4];     // 15 KB combined gates (f16)
  __shared__ float s_qred[NW];

  // ---- one-time weight convert f32 -> f16 into LDS (zero-padded K) ----
  for (int idx = threadIdx.x; idx < 8 * G4 * WROW; idx += TPB) {
    int k = idx % WROW, rem = idx / WROW;
    int n = rem % G4, slot = rem / G4;          // slot = ((b*2+L)*2+m)
    int bb = slot >> 2, LL = (slot >> 1) & 1, m = slot & 1;
    const float* src = (m ? P.bp[bb].wh : P.bp[bb].wi) + LL * (G4 * HH) + n * HH;
    float v = (k < HH) ? src[k] : 0.f;
    ((_Float16*)wbuf)[idx] = (_Float16)v;
  }
  __syncthreads();

  const int  wave  = threadIdx.x >> 5;
  const int  lane  = threadIdx.x & 31;
  const int  gwave = blockIdx.x * NW + wave;
  const int  nw    = gridDim.x * NW;
  const bool lo    = lane < 16;
  const int  k0    = lo ? 0 : 8;
  const int  ncol  = lane & 15;

  _Float16* myh = hbuf[wave][lane];
  float qsum = 0.f;

  for (int b = 0; b < 2; ++b) {
    const BranchP Bp = P.bp[b];
    const float* hx  = b ? P.act_hx0 : P.hx0;
    const float* cxp = b ? P.act_cx0 : P.cx0;

    for (int tile = gwave; tile < NTILES; tile += nw) {
      const int row = tile * 32 + lane;
      const float xv = P.x[row];

      float h[HH];
      { // h0 = tanh(LN20(x*lin1_w + lin1_b))  (unbiased var, eps on std)
        float u[HH]; float mu = 0.f;
        #pragma unroll
        for (int j = 0; j < HH; ++j) { u[j] = xv * Bp.lin1_w[j] + Bp.lin1_b[j]; mu += u[j]; }
        mu *= (1.f / HH);
        float q = 0.f;
        #pragma unroll
        for (int j = 0; j < HH; ++j) { float d = u[j] - mu; q += d * d; }
        float iv = rcp_(sqrt_(q * (1.f / (HH - 1))) + EPSV);
        #pragma unroll
        for (int j = 0; j < HH; ++j)
          h[j] = tanh_((u[j] - mu) * iv * Bp.ln1_g[j] + Bp.ln1_b[j]);
      }

      for (int L = 0; L < NLAYERS; ++L) {
        // h -> LDS (f16, padded to 32), grab both A_h fragments
        #pragma unroll
        for (int j = 0; j < HH; ++j) myh[j] = (_Float16)h[j];
        #pragma unroll
        for (int j = HH; j < WROW; ++j) myh[j] = (_Float16)0.f;
        wave_lds_sync();
        v16h Ah0 = frag_from_lds(hbuf[wave][ncol],      k0);
        v16h Ah1 = frag_from_lds(hbuf[wave][ncol + 16], k0);
        wave_lds_sync();

        const _Float16 (*wiL)[WROW] = &wbuf[(((b * 2 + L) * 2 + 0) * G4)];
        const _Float16 (*whL)[WROW] = &wbuf[(((b * 2 + L) * 2 + 1) * G4)];
        const float* gig = Bp.lgi_g + L * G4;
        const float* gib = Bp.lgi_b + L * G4;
        const float* ghg = Bp.lgh_g + L * G4;
        const float* ghb = Bp.lgh_b + L * G4;

        #pragma unroll
        for (int s = 0; s < 2; ++s) {          // two 16-row sub-tiles
          v16h Ah = s ? Ah1 : Ah0;
          // A_hx straight from global (row-per-lane == same builder as B)
          v16h Ax = frag_from_global(
              hx + ((size_t)L * NPAR + tile * 32 + s * 16 + ncol) * HH, k0, lo);
          v8f Ci[5], Ch[5];
          #pragma unroll
          for (int t = 0; t < 5; ++t) {        // 5 x 16 output columns (4H=80)
            v16h Bi = frag_from_lds(wiL[t * 16 + ncol], k0);
            v16h Bh = frag_from_lds(whL[t * 16 + ncol], k0);
            v8f z = {0.f, 0.f, 0.f, 0.f, 0.f, 0.f, 0.f, 0.f};
            Ci[t] = wmma16(Ah, Bi, z);
            Ch[t] = wmma16(Ax, Bh, z);
          }

          // LN(.,80) per side, per row, directly in C-layout (rows on 16-lane halves)
          v8f Si = Ci[0] + Ci[1] + Ci[2] + Ci[3] + Ci[4];
          v8f Sh = Ch[0] + Ch[1] + Ch[2] + Ch[3] + Ch[4];
          v8f Qi = Ci[0]*Ci[0] + Ci[1]*Ci[1] + Ci[2]*Ci[2] + Ci[3]*Ci[3] + Ci[4]*Ci[4];
          v8f Qh = Ch[0]*Ch[0] + Ch[1]*Ch[1] + Ch[2]*Ch[2] + Ch[3]*Ch[3] + Ch[4]*Ch[4];
          float muI[8], ivI[8], muH[8], ivH[8];
          #pragma unroll
          for (int m = 0; m < 8; ++m) {
            float s1 = hsum16(Si[m]), q1 = hsum16(Qi[m]);
            float mu = s1 * (1.f / G4);
            float va = fmaxf((q1 - (float)G4 * mu * mu) * (1.f / (G4 - 1)), 0.f);
            muI[m] = mu; ivI[m] = rcp_(sqrt_(va) + EPSV);
            float s2 = hsum16(Sh[m]), q2 = hsum16(Qh[m]);
            float mu2 = s2 * (1.f / G4);
            float va2 = fmaxf((q2 - (float)G4 * mu2 * mu2) * (1.f / (G4 - 1)), 0.f);
            muH[m] = mu2; ivH[m] = rcp_(sqrt_(va2) + EPSV);
          }

          const int rbase = s * 16 + (lo ? 0 : 8);
          #pragma unroll
          for (int t = 0; t < 5; ++t) {
            const int n = t * 16 + ncol;
            float g1 = gig[n], b1 = gib[n], g2 = ghg[n], b2 = ghb[n];
            #pragma unroll
            for (int m = 0; m < 8; ++m) {
              float v = (Ci[t][m] - muI[m]) * ivI[m] * g1 + b1
                      + (Ch[t][m] - muH[m]) * ivH[m] * g2 + b2;
              gbuf[wave][rbase + m][n] = (_Float16)v;
            }
          }
        }
        wave_lds_sync();

        // pointwise LSTM cell (lane = row); gates re-read row-major (f16) from LDS
        {
          const float* crow = cxp + ((size_t)L * NPAR + row) * HH;
          const _Float16* grow = gbuf[wave][lane];
          const float* cg = Bp.lnc_g + L * HH;
          const float* cb = Bp.lnc_b + L * HH;
          float cc[HH], og[HH];
          #pragma unroll
          for (int jj = 0; jj < 5; ++jj) {
            v4h ig4 = *(const v4h*)(grow + jj * 4);
            v4h fg4 = *(const v4h*)(grow + 20 + jj * 4);
            v4h og4 = *(const v4h*)(grow + 40 + jj * 4);
            v4h it4 = *(const v4h*)(grow + 60 + jj * 4);
            float4 cx4 = *(const float4*)(crow + jj * 4);
            #pragma unroll
            for (int e = 0; e < 4; ++e) {
              float cxe = (e == 0) ? cx4.x : (e == 1) ? cx4.y : (e == 2) ? cx4.z : cx4.w;
              cc[jj*4+e] = sigmoid_((float)fg4[e] - 1.f) * cxe
                         + sigmoid_((float)ig4[e]) * tanh_((float)it4[e]);
              og[jj*4+e] = (float)og4[e];
            }
          }
          float mu = 0.f;
          #pragma unroll
          for (int j = 0; j < HH; ++j) mu += cc[j];
          mu *= (1.f / HH);
          float q = 0.f;
          #pragma unroll
          for (int j = 0; j < HH; ++j) { float d = cc[j] - mu; q += d * d; }
          float iv = rcp_(sqrt_(q * (1.f / (HH - 1))) + EPSV);
          #pragma unroll
          for (int j = 0; j < HH; ++j)
            h[j] = sigmoid_(og[j]) * tanh_((cc[j] - mu) * iv * cg[j] + cb[j]);
        }
      } // layers

      float o = Bp.out_b[0];
      #pragma unroll
      for (int j = 0; j < HH; ++j) o += h[j] * Bp.out_w[j];
      if (b == 0) P.xout[row] = o;
      else        qsum += sigmoid_(o);
    } // tiles
  } // branches

  // deterministic per-block partial of sum(sigmoid(q_out))
  #pragma unroll
  for (int off = 16; off > 0; off >>= 1) qsum += __shfl_xor(qsum, off, 32);
  if (lane == 0) s_qred[wave] = qsum;
  __syncthreads();
  if (threadIdx.x == 0) {
    float t = 0.f;
    for (int w = 0; w < NW; ++w) t += s_qred[w];
    P.partial[blockIdx.x] = t;
  }
}

__global__ void __launch_bounds__(256)
aml_qreduce(const float* __restrict__ partial, int nb, float* __restrict__ out)
{
  __shared__ float sh[8];
  float s = 0.f;
  for (int i = threadIdx.x; i < nb; i += 256) s += partial[i];
  #pragma unroll
  for (int off = 16; off > 0; off >>= 1) s += __shfl_xor(s, off, 32);
  if ((threadIdx.x & 31) == 0) sh[threadIdx.x >> 5] = s;
  __syncthreads();
  if (threadIdx.x == 0) {
    float t = 0.f;
    #pragma unroll
    for (int w = 0; w < 8; ++w) t += sh[w];
    out[0] = t * (1.f / (float)NPAR);
  }
}

extern "C" void kernel_launch(void* const* d_in, const int* in_sizes, int n_in,
                              void* d_out, int out_size, void* d_ws, size_t ws_size,
                              hipStream_t stream)
{
  (void)in_sizes; (void)n_in; (void)out_size; (void)ws_size;
  KParams P;
  P.x       = (const float*)d_in[0];
  P.hx0     = (const float*)d_in[1];
  P.cx0     = (const float*)d_in[2];
  P.act_hx0 = (const float*)d_in[3];
  P.act_cx0 = (const float*)d_in[4];
  for (int b = 0; b < 2; ++b) {
    const int o = 5 + b * 14;
    BranchP& B = P.bp[b];
    B.lin1_w = (const float*)d_in[o + 0];
    B.lin1_b = (const float*)d_in[o + 1];
    B.ln1_g  = (const float*)d_in[o + 2];
    B.ln1_b  = (const float*)d_in[o + 3];
    B.wi     = (const float*)d_in[o + 4];
    B.wh     = (const float*)d_in[o + 5];
    B.lgi_g  = (const float*)d_in[o + 6];
    B.lgi_b  = (const float*)d_in[o + 7];
    B.lgh_g  = (const float*)d_in[o + 8];
    B.lgh_b  = (const float*)d_in[o + 9];
    B.lnc_g  = (const float*)d_in[o + 10];
    B.lnc_b  = (const float*)d_in[o + 11];
    B.out_w  = (const float*)d_in[o + 12];
    B.out_b  = (const float*)d_in[o + 13];
  }
  P.xout    = (float*)d_out;
  P.partial = (float*)d_ws;

  aml_main<<<NBLK, TPB, 0, stream>>>(P);
  aml_qreduce<<<1, 256, 0, stream>>>((const float*)d_ws, NBLK, (float*)d_out + NPAR);
}
